// DiffusionDet_audio_55714315764091
// MI455X (gfx1250) — 8-bit matrix operand order repaired and checked by reading, compiled, not run on hardware
//
#include <hip/hip_runtime.h>
#include <hip/hip_bf16.h>
#include <math.h>

#define H_HEADS 6
#define BATCH   128
#define NBOX    500
#define NCLS    80
#define NPAD    512
#define IOU_THR 0.5f

typedef int v8i __attribute__((ext_vector_type(8)));

__global__ void keep_kernel(const float* __restrict__ pcls, int* __restrict__ keep) {
  int n = blockIdx.x * blockDim.x + threadIdx.x;
  if (n >= NBOX) return;
  const float* row = pcls + (size_t)(H_HEADS - 1) * BATCH * NBOX * NCLS + (size_t)n * NCLS;
  float m = -3.4e38f;
  for (int c = 0; c < NCLS; ++c) m = fmaxf(m, row[c]);
  keep[n] = (m > 0.0f) ? 1 : 0;
}

__global__ void boxes_kernel(const float* __restrict__ boxes_t,
                             const float* __restrict__ pboxes,
                             const float* __restrict__ noise,
                             const float* __restrict__ fresh,
                             const float* __restrict__ ac,
                             const int* __restrict__ tnow,
                             const int* __restrict__ tnext,
                             const int* __restrict__ keep,
                             float* __restrict__ out) {
  int e = blockIdx.x * blockDim.x + threadIdx.x;
  if (e >= BATCH * NBOX * 2) return;
  float alpha      = ac[tnow[0]];
  float alpha_next = ac[tnext[0]];
  float sqrt_recip   = sqrtf(1.0f / alpha);
  float sqrt_recipm1 = sqrtf(1.0f / alpha - 1.0f);
  float sigma = sqrtf((1.0f - alpha / alpha_next) * (1.0f - alpha_next) / (1.0f - alpha));
  float cc    = sqrtf(1.0f - alpha_next - sigma * sigma);
  float bs  = pboxes[(size_t)(H_HEADS - 1) * BATCH * NBOX * 2 + e];
  float pn  = (sqrt_recip * boxes_t[e] - bs) / sqrt_recipm1;
  float upd = bs * sqrtf(alpha_next) + cc * pn + sigma * noise[e];
  int n = (e >> 1) % NBOX;
  out[e] = keep[n] ? upd : fresh[e];
}

__global__ void softmax_kernel(const float* __restrict__ pcls,
                               float* __restrict__ scores,
                               float* __restrict__ labelsf) {
  int r = blockIdx.x * blockDim.x + threadIdx.x;
  if (r >= BATCH * NBOX) return;
  const float* row = pcls + (size_t)(H_HEADS - 1) * BATCH * NBOX * NCLS + (size_t)r * NCLS;
  float m = -3.4e38f;
  for (int c = 0; c < NCLS; ++c) m = fmaxf(m, row[c]);
  float s = 0.0f, best = -3.4e38f;
  int bi = 0;
  for (int c = 0; c < NCLS; ++c) {
    float x = row[c];
    s += expf(x - m);
    if (c < NCLS - 1 && x > best) { best = x; bi = c; }
  }
  scores[r]  = expf(best - m) / s;
  labelsf[r] = (float)bi;
}

__global__ __launch_bounds__(256) void nms_kernel(const float* __restrict__ pboxes,
                                                  const float* __restrict__ scores,
                                                  const float* __restrict__ labelsf,
                                                  float* __restrict__ keep_out) {
  __shared__ float        key[NPAD];
  __shared__ int          idxs[NPAD];
  __shared__ float        x1s[NPAD], x2s[NPAD];
  __shared__ int          labs[NPAD];
  __shared__ unsigned int M[NPAD * 16];
  __shared__ unsigned int sup[16];

  int b   = blockIdx.x;
  int tid = threadIdx.x;
  const float* sc = scores  + (size_t)b * NBOX;
  const float* lf = labelsf + (size_t)b * NBOX;
  const float* pb = pboxes + (size_t)(H_HEADS - 1) * BATCH * NBOX * 2 + (size_t)b * NBOX * 2;

  for (int i = tid; i < NPAD; i += 256) {
    key[i]  = (i < NBOX) ? sc[i] : -3.4e38f;
    idxs[i] = i;
  }
  for (int i = tid; i < NPAD * 16; i += 256) M[i] = 0u;

  for (int k = 2; k <= NPAD; k <<= 1) {
    for (int j = k >> 1; j > 0; j >>= 1) {
      __syncthreads();
      for (int i = tid; i < NPAD; i += 256) {
        int l = i ^ j;
        if (l > i) {
          float ki = key[i], kl = key[l];
          int   ii = idxs[i], il = idxs[l];
          bool after = (ki < kl) || (ki == kl && ii > il);
          bool up = (i & k) == 0;
          if (up == after) { key[i] = kl; key[l] = ki; idxs[i] = il; idxs[l] = ii; }
        }
      }
    }
  }
  __syncthreads();

  for (int i = tid; i < NPAD; i += 256) {
    int o = idxs[i];
    if (o < NBOX) {
      float c = pb[o * 2 + 0];
      float w = fmaxf(pb[o * 2 + 1], 0.0001f);
      x1s[i] = c - 0.5f * w;
      x2s[i] = c + 0.5f * w;
      labs[i] = (int)(lf[o] + 0.5f);
    } else {
      x1s[i] = 0.0f; x2s[i] = 0.0f; labs[i] = -1;
    }
  }
  __syncthreads();

  int wave = tid >> 5, lane = tid & 31;
  int half = (lane >> 4) & 1;
  for (int t = wave; t < (NPAD / 16) * (NPAD / 16); t += 8) {
    int ti = t >> 5, tj = t & 31;
    if (tj < ti) continue;
    int la = labs[ti * 16 + (lane & 15)];
    int lb = labs[tj * 16 + (lane & 15)];
    int byteA = 1 << ((la & 3) << 3);
    int byteB = 1 << ((lb & 3) << 3);
    v8i acc;
#pragma unroll
    for (int z = 0; z < 8; ++z) acc[z] = 0;
#pragma unroll
    for (int pass = 0; pass < 2; ++pass) {
      int K0 = pass << 6;
      int sA = (la - K0) >> 2;
      int sB = (lb - K0) >> 2;
      v8i A, Bm;
#pragma unroll
      for (int v = 0; v < 8; ++v) {
        int slotA = ((v >> 1) << 2) + (v & 1) + (half << 1);
        A[v] = (sA == slotA) ? byteA : 0;
        int slotB = ((v >> 1) << 2) + (v & 1) + (half << 1);
        Bm[v] = (sB == slotB) ? byteB : 0;
      }
      acc = __builtin_amdgcn_wmma_i32_16x16x64_iu8(false, A, false, Bm, acc, false, false);
    }
#pragma unroll
    for (int d = 0; d < 8; ++d) {
      if (acc[d] != 0) {
        int row = ti * 16 + d + (half << 3);
        int col = tj * 16 + (lane & 15);
        atomicOr(&M[row * 16 + (col >> 5)], 1u << (col & 31));
      }
    }
  }
  __syncthreads();

  for (int task = tid; task < NPAD * 16; task += 256) {
    int i    = task >> 4;
    int base = (task & 15) << 5;
    int s = i - base;
    unsigned int gt = (s < 0) ? 0xffffffffu : ((s >= 31) ? 0u : (0xffffffffu << (s + 1)));
    unsigned int u = M[task] & gt;
    unsigned int bits = 0u;
    if (u) {
      float xi1 = x1s[i], xi2 = x2s[i];
      float wi  = xi2 - xi1;
      do {
        int bb = __builtin_ctz(u);
        u &= u - 1u;
        int j = base + bb;
        float inter = fmaxf(0.0f, fminf(xi2, x2s[j]) - fmaxf(xi1, x1s[j]));
        float uni   = wi + (x2s[j] - x1s[j]) - inter;
        if (inter > IOU_THR * fmaxf(uni, 1e-9f)) bits |= 1u << bb;
      } while (u);
    }
    M[task] = bits;
  }
  __syncthreads();

  if (tid < 32) {
    unsigned int supw = 0u;
    for (int i = 0; i < NBOX; ++i) {
      unsigned int row = (tid < 16) ? M[i * 16 + tid] : 0u;
      unsigned int w = __shfl(supw, i >> 5, 32);
      if (((w >> (i & 31)) & 1u) == 0u) supw |= row;
    }
    if (tid < 16) sup[tid] = supw;
  }
  __syncthreads();

  for (int i = tid; i < NBOX; i += 256) {
    bool kept = ((sup[i >> 5] >> (i & 31)) & 1u) == 0u;
    keep_out[(size_t)b * NBOX + idxs[i]] = kept ? 1.0f : 0.0f;
  }
}

extern "C" void kernel_launch(void* const* d_in, const int* in_sizes, int n_in,
                              void* d_out, int out_size, void* d_ws, size_t ws_size,
                              hipStream_t stream) {
  (void)in_sizes; (void)n_in; (void)out_size; (void)ws_size;
  const float* boxes_t = (const float*)d_in[0];
  const float* pcls    = (const float*)d_in[1];
  const float* pboxes  = (const float*)d_in[2];
  const float* noise   = (const float*)d_in[3];
  const float* fresh   = (const float*)d_in[4];
  const float* ac      = (const float*)d_in[5];
  const int*   tnow    = (const int*)d_in[6];
  const int*   tnext   = (const int*)d_in[7];

  float* out        = (float*)d_out;
  float* boxes_next = out;
  float* scores     = out + 128000;
  float* labelsf    = out + 192000;
  float* nmskeep    = out + 256000;
  int*   keep       = (int*)d_ws;

  keep_kernel   <<<2, 256, 0, stream>>>(pcls, keep);
  boxes_kernel  <<<(BATCH * NBOX * 2 + 255) / 256, 256, 0, stream>>>(
      boxes_t, pboxes, noise, fresh, ac, tnow, tnext, keep, boxes_next);
  softmax_kernel<<<(BATCH * NBOX + 255) / 256, 256, 0, stream>>>(pcls, scores, labelsf);
  nms_kernel    <<<BATCH, 256, 0, stream>>>(pboxes, scores, labelsf, nmskeep);
}
